// MTKDecoderKVCacheV2_18957985644738
// MI455X (gfx1250) — compile-verified, matrix operand order checked by reading, not run on hardware
//
#include <hip/hip_runtime.h>
#include <hip/hip_bf16.h>

#define LAYERS 6
#define BATCH 16
#define DMODEL 1024
#define NHEAD 16
#define HEADDIM 64
#define FFNDIM 4096
#define VOCAB 32000
#define NCACHE 512
#define NSRC 512

typedef __attribute__((ext_vector_type(16))) __bf16 v16bf;
typedef __attribute__((ext_vector_type(2)))  __bf16 v2bf;
typedef __attribute__((ext_vector_type(8)))  float  v8f;

union Frag { unsigned int u[8]; v16bf bf; };

// hardware fp32->bf16 convert (v_cvt_pk_bf16_f32), packed pair into one dword
__device__ __forceinline__ unsigned int pack2_bf16(float a, float b) {
  v2bf c;
  c.x = (__bf16)a;
  c.y = (__bf16)b;
  return __builtin_bit_cast(unsigned int, c);
}

// Fragment from an fp32 row (row pointer p = base + row*ld + k0).
// Per ISA 7.12.2 A/B layout: lane<16 holds K {0..7,16..23}, lane>=16 holds
// K {8..15,24..31} (khalf = (lane>>4)*8); two K per VGPR, low K in [15:0].
__device__ __forceinline__ void load_frag_f32(Frag& f, const float* __restrict__ p,
                                              int khalf) {
  const float4 t0 = *(const float4*)(p + khalf);
  const float4 t1 = *(const float4*)(p + khalf + 4);
  const float4 t2 = *(const float4*)(p + khalf + 16);
  const float4 t3 = *(const float4*)(p + khalf + 20);
  f.u[0] = pack2_bf16(t0.x, t0.y); f.u[1] = pack2_bf16(t0.z, t0.w);
  f.u[2] = pack2_bf16(t1.x, t1.y); f.u[3] = pack2_bf16(t1.z, t1.w);
  f.u[4] = pack2_bf16(t2.x, t2.y); f.u[5] = pack2_bf16(t2.z, t2.w);
  f.u[6] = pack2_bf16(t3.x, t3.y); f.u[7] = pack2_bf16(t3.z, t3.w);
}

// Fragment from a bf16 row: two contiguous 16B runs -> two 128b loads, no VALU.
// Works for both global and LDS pointers (addrspace inferred by the compiler).
template <typename PTR>
__device__ __forceinline__ void load_frag_bf16(Frag& f, PTR p, int khalf) {
  const uint4 t0 = *(const uint4*)(p + khalf);        // K khalf .. khalf+7
  const uint4 t1 = *(const uint4*)(p + khalf + 16);   // K khalf+16 .. khalf+23
  f.u[0] = t0.x; f.u[1] = t0.y; f.u[2] = t0.z; f.u[3] = t0.w;
  f.u[4] = t1.x; f.u[5] = t1.y; f.u[6] = t1.z; f.u[7] = t1.w;
}

// CDNA5 async global->LDS copy, 16B per lane. Same IOFFSET applies to both
// addresses (ISA 15.18.3). LDS flat address low 32 bits == LDS offset.
template <int OFF>
__device__ __forceinline__ void async_copy16(unsigned lds_off,
                                             const unsigned short* g) {
  asm volatile("global_load_async_to_lds_b128 %0, %1, off offset:%2"
               :: "v"(lds_off), "v"(g), "n"(OFF) : "memory");
}
__device__ __forceinline__ void wait_async0() {
  asm volatile("s_wait_asynccnt 0" ::: "memory");
}
// stage one 64B row slice (32 bf16) of the A slab
__device__ __forceinline__ void stage_row(unsigned lds_off, const unsigned short* g) {
  async_copy16<0>(lds_off, g);
  async_copy16<16>(lds_off, g);
  async_copy16<32>(lds_off, g);
  async_copy16<48>(lds_off, g);
}

// fp32 -> bf16 bulk convert (4 elems/thread), n multiple of 4
__global__ void cvt_bf16(const float* __restrict__ src, unsigned short* __restrict__ dst,
                         size_t n) {
  size_t i = ((size_t)blockIdx.x * 256 + threadIdx.x) * 4;
  if (i >= n) return;
  const float4 t = *(const float4*)(src + i);
  uint2 o;
  o.x = pack2_bf16(t.x, t.y);
  o.y = pack2_bf16(t.z, t.w);
  *(uint2*)(dst + i) = o;
}

// y[16,N] = x[16,K] * w[N,K]^T + bias ; act==1 -> SiLU. One 16x16 tile per wave.
// Weight-bandwidth bound (each w element touched exactly once).
__global__ void gemm_mb16(const float* __restrict__ x, const float* __restrict__ w,
                          const float* __restrict__ bias, float* __restrict__ y,
                          int K, int ldy, int act) {
  const int lane  = threadIdx.x & 31;
  const int wv    = threadIdx.x >> 5;
  const int ntile = (blockIdx.x * 4 + wv) * 16;
  const int r     = lane & 15;
  const int khalf = (lane >> 4) << 3;
  const float* xp = x + (size_t)r * K;
  const float* wp = w + (size_t)(ntile + r) * K;
  v8f acc = {0.f, 0.f, 0.f, 0.f, 0.f, 0.f, 0.f, 0.f};
  for (int k0 = 0; k0 < K; k0 += 32) {
    Frag a, b;
    load_frag_f32(a, xp + k0, khalf);
    load_frag_f32(b, wp + k0, khalf);
    acc = __builtin_amdgcn_wmma_f32_16x16x32_bf16(false, a.bf, false, b.bf,
                                                  (short)0, acc, false, false);
  }
  const int n     = ntile + (lane & 15);
  const int mbase = (lane >> 4) << 3;
  const float bv  = bias ? bias[n] : 0.f;
#pragma unroll
  for (int rr = 0; rr < 8; ++rr) {
    float v = acc[rr] + bv;
    if (act == 1) v = v * __builtin_amdgcn_rcpf(1.f + __expf(-v));
    y[(size_t)(mbase + rr) * ldy + n] = v;
  }
}

#define LDSROW 40   // 32 bf16 data + 8 pad -> 80B row stride, bank-conflict free

// Y[M,N] = A[M,K](bf16) * W[N,K](bf16)^T + bias.
// Block = 128 threads (4 waves), tile 128(M) x 128(N).
// A slab (128x32 bf16, 8KB) double-buffered in LDS: chunk k+1 is prefetched
// via async global->LDS while the 16 WMMAs of chunk k execute; one barrier
// per chunk. Each wave computes 8(M) x 2(N) tiles, reusing every A fragment
// for two WMMAs.
__global__ void gemm_big(const unsigned short* __restrict__ A,
                         const unsigned short* __restrict__ W,
                         const float* __restrict__ bias, float* __restrict__ Y,
                         int K, int N) {
  __shared__ unsigned short Atile[2][128 * LDSROW];
  const int tid   = threadIdx.x;
  const int lane  = tid & 31;
  const int wv    = tid >> 5;
  const int mtile = blockIdx.x * 128;
  const int n0    = blockIdx.y * 128 + wv * 32;       // 2 n-tiles: n0, n0+16
  const int r     = lane & 15;
  const int khalf = (lane >> 4) << 3;

  const unsigned short* arow = A + (size_t)(mtile + tid) * K;  // stage source
  const unsigned lrow[2] = { (unsigned)(size_t)&Atile[0][tid * LDSROW],
                             (unsigned)(size_t)&Atile[1][tid * LDSROW] };
  const unsigned short* wp0 = W + (size_t)(n0 + r) * K;
  const unsigned short* wp1 = wp0 + (size_t)16 * K;

  v8f acc[2][8];
#pragma unroll
  for (int j = 0; j < 2; ++j)
#pragma unroll
    for (int i = 0; i < 8; ++i)
      acc[j][i] = (v8f){0.f, 0.f, 0.f, 0.f, 0.f, 0.f, 0.f, 0.f};

  // prologue: stage chunk 0
  stage_row(lrow[0], arow);
  wait_async0();
  __syncthreads();

  for (int k0 = 0; k0 < K; k0 += 32) {
    const int buf = (k0 >> 5) & 1;
    // prefetch next chunk into the other buffer (its readers finished last iter)
    if (k0 + 32 < K) stage_row(lrow[buf ^ 1], arow + k0 + 32);
    Frag b0, b1;
    load_frag_bf16(b0, wp0 + k0, khalf);
    load_frag_bf16(b1, wp1 + k0, khalf);
#pragma unroll
    for (int i = 0; i < 8; ++i) {
      Frag a;
      load_frag_bf16(a, (const unsigned short*)&Atile[buf][(i * 16 + r) * LDSROW], khalf);
      acc[0][i] = __builtin_amdgcn_wmma_f32_16x16x32_bf16(false, a.bf, false, b0.bf,
                                                          (short)0, acc[0][i], false, false);
      acc[1][i] = __builtin_amdgcn_wmma_f32_16x16x32_bf16(false, a.bf, false, b1.bf,
                                                          (short)0, acc[1][i], false, false);
    }
    wait_async0();        // next chunk's slab complete (own wave's copies)
    __syncthreads();      // visible to all waves; frees buffer we just read
  }

  const int mbase = mtile + ((lane >> 4) << 3);
#pragma unroll
  for (int j = 0; j < 2; ++j) {
    const int n    = n0 + j * 16 + (lane & 15);
    const float bv = bias ? bias[n] : 0.f;
#pragma unroll
    for (int i = 0; i < 8; ++i)
#pragma unroll
      for (int rr = 0; rr < 8; ++rr)
        Y[(size_t)(mbase + i * 16 + rr) * N + n] = acc[j][i][rr] + bv;
  }
}

// softmax attention for one (b,h): keys = ncached cached rows (+1 new row if nk)
__global__ void attention(const float* __restrict__ q, const float* __restrict__ pk,
                          const float* __restrict__ pv, const float* __restrict__ nk,
                          const float* __restrict__ nv, float* __restrict__ o,
                          int ncached) {
  const int b = blockIdx.x >> 4;
  const int h = blockIdx.x & 15;
  const int t = threadIdx.x;
  const int nkeys = ncached + (nk ? 1 : 0);
  __shared__ float sq[HEADDIM];
  __shared__ float sc[NCACHE + 1];
  __shared__ float red[256];
  if (t < HEADDIM) sq[t] = q[b * DMODEL + h * HEADDIM + t] * 0.125f; // 1/sqrt(64)
  __syncthreads();
  for (int k = t; k < nkeys; k += 256) {
    const float* kp = (k < ncached)
        ? (pk + ((size_t)b * ncached + k) * DMODEL + h * HEADDIM)
        : (nk + (size_t)b * DMODEL + h * HEADDIM);
    float s = 0.f;
#pragma unroll
    for (int d4 = 0; d4 < 16; ++d4) {
      const float4 kk = *(const float4*)(kp + d4 * 4);
      s += sq[d4 * 4] * kk.x + sq[d4 * 4 + 1] * kk.y +
           sq[d4 * 4 + 2] * kk.z + sq[d4 * 4 + 3] * kk.w;
    }
    sc[k] = s;
  }
  __syncthreads();
  float m = -1e30f;
  for (int k = t; k < nkeys; k += 256) m = fmaxf(m, sc[k]);
  red[t] = m; __syncthreads();
  for (int s = 128; s > 0; s >>= 1) { if (t < s) red[t] = fmaxf(red[t], red[t + s]); __syncthreads(); }
  m = red[0]; __syncthreads();
  float sum = 0.f;
  for (int k = t; k < nkeys; k += 256) { float e = __expf(sc[k] - m); sc[k] = e; sum += e; }
  red[t] = sum; __syncthreads();
  for (int s = 128; s > 0; s >>= 1) { if (t < s) red[t] += red[t + s]; __syncthreads(); }
  const float inv = 1.f / red[0];
  __syncthreads();
  const int d = t & 63, c = t >> 6;
  float acc = 0.f;
  for (int k = c; k < nkeys; k += 4) {
    const float* vp = (k < ncached)
        ? (pv + ((size_t)b * ncached + k) * DMODEL + h * HEADDIM)
        : (nv + (size_t)b * DMODEL + h * HEADDIM);
    acc += sc[k] * vp[d];
  }
  red[t] = acc; __syncthreads();
  if (c == 0)
    o[b * DMODEL + h * HEADDIM + d] =
        (red[d] + red[64 + d] + red[128 + d] + red[192 + d]) * inv;
}

// h = LayerNorm(res + y) * g + be   (one block per batch row, in-place safe)
__global__ void add_ln(const float* __restrict__ res, const float* __restrict__ y,
                       const float* __restrict__ g, const float* __restrict__ be,
                       float* __restrict__ h) {
  const int b = blockIdx.x, t = threadIdx.x;
  __shared__ float red[256];
  float v[4]; float s = 0.f;
#pragma unroll
  for (int j = 0; j < 4; ++j) {
    int d = t + j * 256;
    v[j] = res[b * DMODEL + d] + y[b * DMODEL + d];
    s += v[j];
  }
  red[t] = s; __syncthreads();
  for (int st = 128; st > 0; st >>= 1) { if (t < st) red[t] += red[t + st]; __syncthreads(); }
  const float mean = red[0] * (1.f / DMODEL);
  __syncthreads();
  s = 0.f;
#pragma unroll
  for (int j = 0; j < 4; ++j) { float d0 = v[j] - mean; s += d0 * d0; }
  red[t] = s; __syncthreads();
  for (int st = 128; st > 0; st >>= 1) { if (t < st) red[t] += red[t + st]; __syncthreads(); }
  const float rstd = rsqrtf(red[0] * (1.f / DMODEL) + 1e-5f);
  __syncthreads();
#pragma unroll
  for (int j = 0; j < 4; ++j) {
    int d = t + j * 256;
    h[b * DMODEL + d] = (v[j] - mean) * rstd * g[d] + be[d];
  }
}

// h = decoder_embed * sqrt(D) + pe[cache_len[0]]
__global__ void embed_pe(const float* __restrict__ de, const float* __restrict__ pe,
                         const int* __restrict__ clen, float* __restrict__ h) {
  const int i = blockIdx.x * 256 + threadIdx.x;   // 0..B*D-1
  const int d = i & (DMODEL - 1);
  h[i] = de[i] * 32.0f + pe[(size_t)clen[0] * DMODEL + d];
}

extern "C" void kernel_launch(void* const* d_in, const int* in_sizes, int n_in,
                              void* d_out, int out_size, void* d_ws, size_t ws_size,
                              hipStream_t stream) {
  const float* dec   = (const float*)d_in[0];
  const float* enc   = (const float*)d_in[1];
  const float* pkeys = (const float*)d_in[2];
  const float* pvals = (const float*)d_in[3];
  const int*   clen  = (const int*)d_in[4];
  const float* pe    = (const float*)d_in[5];
  const float* sa_qw = (const float*)d_in[6];
  const float* sa_kw = (const float*)d_in[7];
  const float* sa_vw = (const float*)d_in[8];
  const float* sa_ow = (const float*)d_in[9];
  const float* ca_qw = (const float*)d_in[10];
  const float* ca_kw = (const float*)d_in[11];
  const float* ca_vw = (const float*)d_in[12];
  const float* ca_ow = (const float*)d_in[13];
  const float* sa_qb = (const float*)d_in[14];
  const float* sa_kb = (const float*)d_in[15];
  const float* sa_vb = (const float*)d_in[16];
  const float* sa_ob = (const float*)d_in[17];
  const float* ca_qb = (const float*)d_in[18];
  const float* ca_kb = (const float*)d_in[19];
  const float* ca_vb = (const float*)d_in[20];
  const float* ca_ob = (const float*)d_in[21];
  const float* fc1_w = (const float*)d_in[22];
  const float* fc1_b = (const float*)d_in[23];
  const float* fc2_w = (const float*)d_in[24];
  const float* fc2_b = (const float*)d_in[25];
  const float* ln1_w = (const float*)d_in[26];
  const float* ln2_w = (const float*)d_in[27];
  const float* ln3_w = (const float*)d_in[28];
  const float* ln1_b = (const float*)d_in[29];
  const float* ln2_b = (const float*)d_in[30];
  const float* ln3_b = (const float*)d_in[31];
  const float* lm_w  = (const float*)d_in[32];

  float* out = (float*)d_out;
  float* ws  = (float*)d_ws;

  // workspace layout (fp32 elements, then bf16 region)
  float* h    = ws;                 // [16,1024]
  float* q    = ws + 16384;         // [16,1024]
  float* attn = ws + 32768;         // [16,1024]
  float* y    = ws + 49152;         // [16,1024]
  float* ffn1 = ws + 65536;         // [16,4096]
  float* ek   = ws + 131072;        // [16,512,1024]
  float* ev   = ek + (size_t)BATCH * NSRC * DMODEL;
  unsigned short* enc_bf = (unsigned short*)(ev + (size_t)BATCH * NSRC * DMODEL);
  unsigned short* wk_bf  = enc_bf + (size_t)BATCH * NSRC * DMODEL;   // [1024,1024]
  unsigned short* wv_bf  = wk_bf + (size_t)DMODEL * DMODEL;          // [1024,1024]

  float* logits = out;                                     // [16,1,32000]
  float* new_k  = out + (size_t)BATCH * VOCAB;             // [6,16,1,1024]
  float* new_v  = new_k + (size_t)LAYERS * BATCH * DMODEL; // [6,16,1,1024]

  embed_pe<<<(BATCH * DMODEL) / 256, 256, 0, stream>>>(dec, pe, clen, h);

  // encoder states -> bf16 once (reused by all 12 big GEMMs)
  const size_t nEnc = (size_t)BATCH * NSRC * DMODEL;
  cvt_bf16<<<(unsigned)(nEnc / 4 / 256), 256, 0, stream>>>(enc, enc_bf, nEnc);

  const dim3 gD(DMODEL / 64);      // 16 blocks x 4 waves -> 64 N-tiles
  const dim3 gF(FFNDIM / 64);
  const dim3 gBig(BATCH * NSRC / 128, DMODEL / 128);
  const size_t nW = (size_t)DMODEL * DMODEL;
  const unsigned gW = (unsigned)(nW / 4 / 256);

  for (int l = 0; l < LAYERS; ++l) {
    const size_t wo = (size_t)l * DMODEL * DMODEL;
    const size_t bo = (size_t)l * DMODEL;
    float* nk = new_k + (size_t)l * BATCH * DMODEL;
    float* nv = new_v + (size_t)l * BATCH * DMODEL;

    // ---- self-attention ----
    gemm_mb16<<<gD, 128, 0, stream>>>(h, sa_qw + wo, sa_qb + bo, q,  DMODEL, DMODEL, 0);
    gemm_mb16<<<gD, 128, 0, stream>>>(h, sa_kw + wo, sa_kb + bo, nk, DMODEL, DMODEL, 0);
    gemm_mb16<<<gD, 128, 0, stream>>>(h, sa_vw + wo, sa_vb + bo, nv, DMODEL, DMODEL, 0);
    attention<<<BATCH * NHEAD, 256, 0, stream>>>(
        q, pkeys + (size_t)l * BATCH * NCACHE * DMODEL,
        pvals + (size_t)l * BATCH * NCACHE * DMODEL, nk, nv, attn, NCACHE);
    gemm_mb16<<<gD, 128, 0, stream>>>(attn, sa_ow + wo, sa_ob + bo, y, DMODEL, DMODEL, 0);
    add_ln<<<BATCH, 256, 0, stream>>>(h, y, ln1_w + bo, ln1_b + bo, h);

    // ---- cross-attention ----
    gemm_mb16<<<gD, 128, 0, stream>>>(h, ca_qw + wo, ca_qb + bo, q, DMODEL, DMODEL, 0);
    cvt_bf16<<<gW, 256, 0, stream>>>(ca_kw + wo, wk_bf, nW);
    cvt_bf16<<<gW, 256, 0, stream>>>(ca_vw + wo, wv_bf, nW);
    gemm_big<<<gBig, 128, 0, stream>>>(enc_bf, wk_bf, ca_kb + bo, ek, DMODEL, DMODEL);
    gemm_big<<<gBig, 128, 0, stream>>>(enc_bf, wv_bf, ca_vb + bo, ev, DMODEL, DMODEL);
    attention<<<BATCH * NHEAD, 256, 0, stream>>>(q, ek, ev, nullptr, nullptr, attn, NSRC);
    gemm_mb16<<<gD, 128, 0, stream>>>(attn, ca_ow + wo, ca_ob + bo, y, DMODEL, DMODEL, 0);
    add_ln<<<BATCH, 256, 0, stream>>>(h, y, ln2_w + bo, ln2_b + bo, h);

    // ---- FFN (SiLU) ----
    gemm_mb16<<<gF, 128, 0, stream>>>(h, fc1_w + (size_t)l * FFNDIM * DMODEL,
                                      fc1_b + (size_t)l * FFNDIM, ffn1, DMODEL, FFNDIM, 1);
    gemm_mb16<<<gD, 128, 0, stream>>>(ffn1, fc2_w + (size_t)l * DMODEL * FFNDIM,
                                      fc2_b + bo, y, FFNDIM, DMODEL, 0);
    add_ln<<<BATCH, 256, 0, stream>>>(h, y, ln3_w + bo, ln3_b + bo, h);
  }

  // ---- LM head (no bias) ----
  gemm_mb16<<<VOCAB / 64, 128, 0, stream>>>(h, lm_w, nullptr, logits, DMODEL, VOCAB, 0);
}
